// BiLSTM_67250597921508
// MI455X (gfx1250) — compile-verified, matrix operand order checked by reading, not run on hardware
//
#include <hip/hip_runtime.h>

// ---------------- problem constants ----------------
#define B_      64
#define T_      256
#define H_      300
#define HP_     320     // H padded to mult of 32 (WMMA K)
#define LDA_    640     // layer-input leading dim (2*HP_), covers concat 600
#define GP_     1280    // 4 gates * HP_
#define K_      11
#define VSTART_ 9
#define VSTOP_  10

typedef _Float16 f16_t;
typedef __attribute__((ext_vector_type(16))) _Float16 f16x16;
typedef __attribute__((ext_vector_type(8)))  float    f32x8;

union TileU { uint4 q[2]; f16x16 v; };

// Per-lane 16x32 f16 operand tile load (row-major source, ld in elements).
// ISA layout: lanes 0-15 -> row=lane, K = [0..7] and [16..23]
//             lanes 16-31 -> row=lane-16, K = [8..15] and [24..31]
__device__ __forceinline__ f16x16 load_tile16(const f16_t* p0, int ld, int row0,
                                              int k0, int lane) {
  const int r  = lane & 15;
  const int kb = (lane & 16) ? 8 : 0;
  const f16_t* p = p0 + (size_t)(row0 + r) * ld + (k0 + kb);
  TileU t;
  t.q[0] = *(const uint4*)(p);
  t.q[1] = *(const uint4*)(p + 16);
  return t.v;
}

// ---------------- elementwise prep kernels ----------------
__global__ void k_gather_embed(const int* __restrict__ x, const float* __restrict__ emb,
                               f16_t* __restrict__ out) {
  int idx = blockIdx.x * blockDim.x + threadIdx.x;
  if (idx >= B_ * T_ * LDA_) return;
  int c  = idx % LDA_;
  int bt = idx / LDA_;
  float v = 0.f;
  if (c < H_) v = emb[(size_t)x[bt] * H_ + c];
  out[idx] = (f16_t)v;
}

__global__ void k_prep_wih(const float* __restrict__ W, f16_t* __restrict__ out, int in_dim) {
  int idx = blockIdx.x * blockDim.x + threadIdx.x;
  if (idx >= GP_ * LDA_) return;
  int k = idx % LDA_, row = idx / LDA_;
  int g = row / HP_,  j   = row % HP_;
  float v = 0.f;
  if (j < H_ && k < in_dim) v = W[(size_t)(g * H_ + j) * in_dim + k];
  out[idx] = (f16_t)v;
}

__global__ void k_prep_whh(const float* __restrict__ W, f16_t* __restrict__ out) {
  int idx = blockIdx.x * blockDim.x + threadIdx.x;
  if (idx >= GP_ * HP_) return;
  int k = idx % HP_, row = idx / HP_;
  int g = row / HP_, j   = row % HP_;
  float v = 0.f;
  if (j < H_ && k < H_) v = W[(size_t)(g * H_ + j) * H_ + k];
  out[idx] = (f16_t)v;
}

__global__ void k_prep_bias(const float* __restrict__ b, float* __restrict__ out) {
  int idx = blockIdx.x * blockDim.x + threadIdx.x;
  if (idx >= GP_) return;
  int g = idx / HP_, j = idx % HP_;
  out[idx] = (j < H_) ? b[g * H_ + j] : 0.f;
}

__global__ void k_rev(const f16_t* __restrict__ in, f16_t* __restrict__ out,
                      const int* __restrict__ lens) {
  int idx = blockIdx.x * blockDim.x + threadIdx.x;
  if (idx >= B_ * T_ * LDA_) return;
  int c = idx % LDA_;
  int t = (idx / LDA_) % T_;
  int b = idx / (LDA_ * T_);
  int len = lens[b];
  int src = (t < len) ? (len - 1 - t) : t;
  out[idx] = in[((size_t)b * T_ + src) * LDA_ + c];
}

__global__ void k_concat(const f16_t* __restrict__ hf, const f16_t* __restrict__ hb,
                         const int* __restrict__ lens, f16_t* __restrict__ out) {
  int idx = blockIdx.x * blockDim.x + threadIdx.x;
  if (idx >= B_ * T_ * LDA_) return;
  int c = idx % LDA_;
  int t = (idx / LDA_) % T_;
  int b = idx / (LDA_ * T_);
  int len = lens[b];
  float m = (t < len) ? 1.f : 0.f;
  float v = 0.f;
  if (c < H_) {
    v = (float)hf[((size_t)b * T_ + t) * HP_ + c];
  } else if (c < 2 * H_) {
    int src = (t < len) ? (len - 1 - t) : t;
    v = (float)hb[((size_t)b * T_ + src) * HP_ + (c - H_)];
  }
  out[idx] = (f16_t)(v * m);
}

// ---------------- input-projection GEMM (WMMA, ping-pong pipelined) --------
// X: [B_*T_ rows, ld LDA_] f16 ; W: [GP_ rows, ld LDA_] f16 (row n = output col n)
// XW out layout: [T_][B_][GP_] f32
// kdim is always a multiple of 64 (320 or 640) -> even number of K-tiles, so a
// 2x-unrolled ping-pong needs no register rotation (no v_mov copies).
__global__ void __launch_bounds__(256)
k_gemm_xw(const f16_t* __restrict__ X, int kdim, const f16_t* __restrict__ W,
          const float* __restrict__ bias, float* __restrict__ XW) {
  const int lane = threadIdx.x & 31;
  const int wave = threadIdx.x >> 5;
  const int job  = blockIdx.x * 8 + wave;    // 1024 Mtiles * 20 Ngroups = 20480
  const int mt   = job / 20;
  const int ng   = job % 20;
  const int row0 = mt * 16;
  f32x8 acc[4] = {};
  // prologue: load K-tile 0 into ping set
  f16x16 aP = load_tile16(X, LDA_, row0, 0, lane);
  f16x16 bP[4];
#pragma unroll
  for (int j = 0; j < 4; ++j) bP[j] = load_tile16(W, LDA_, ng * 64 + j * 16, 0, lane);
  for (int k0 = 0; k0 < kdim; k0 += 64) {
    const int k1 = k0 + 32;                              // always < kdim
    const int k2 = (k0 + 64 < kdim) ? (k0 + 64) : 0;     // last iter: dummy reload
    if (k0 + 64 < kdim) {
      __builtin_prefetch(X + (size_t)(row0 + (lane & 15)) * LDA_ + k0 + 64, 0, 3);
      __builtin_prefetch(W + (size_t)(ng * 64 + (lane & 15)) * LDA_ + k0 + 64, 0, 3);
    }
    // load pong set (k1) while computing on ping set (k0)
    f16x16 aQ = load_tile16(X, LDA_, row0, k1, lane);
    f16x16 bQ[4];
#pragma unroll
    for (int j = 0; j < 4; ++j) bQ[j] = load_tile16(W, LDA_, ng * 64 + j * 16, k1, lane);
#pragma unroll
    for (int j = 0; j < 4; ++j)
      acc[j] = __builtin_amdgcn_wmma_f32_16x16x32_f16(false, aP, false, bP[j],
                                                      (short)0, acc[j], false, false);
    // load ping set (k2) while computing on pong set (k1)
    aP = load_tile16(X, LDA_, row0, k2, lane);
#pragma unroll
    for (int j = 0; j < 4; ++j) bP[j] = load_tile16(W, LDA_, ng * 64 + j * 16, k2, lane);
#pragma unroll
    for (int j = 0; j < 4; ++j)
      acc[j] = __builtin_amdgcn_wmma_f32_16x16x32_f16(false, aQ, false, bQ[j],
                                                      (short)0, acc[j], false, false);
  }
  const int n_l = lane & 15;
  const int mb  = row0 + ((lane >> 4) << 3);
#pragma unroll
  for (int j = 0; j < 4; ++j) {
    int col  = ng * 64 + j * 16 + n_l;
    float bv = bias[col];
#pragma unroll
    for (int r = 0; r < 8; ++r) {
      int row = mb + r;
      int bb  = row >> 8;     // / T_
      int tt  = row & 255;    // % T_
      XW[((size_t)tt * B_ + bb) * GP_ + col] = acc[j][r] + bv;
    }
  }
}

// ---------------- recurrent scan (persistent single workgroup, WMMA) ----------------
// XW: [T_][B_][GP_] f32 ; Whh: [GP_][HP_] f16 ; cst: [B_*HP_] f32 scratch ; h_all: [B_][T_][HP_] f16
// 512 threads = 16 waves; 80 (Mtile,Jtile) pairs -> exactly 5 pairs per wave.
__global__ void __launch_bounds__(512)
k_lstm_scan(const float* __restrict__ XW, const f16_t* __restrict__ Whh,
            float* __restrict__ cst, f16_t* __restrict__ h_all) {
  __shared__ f16_t hbuf[B_ * HP_];     // 40 KB
  const int tid  = threadIdx.x;
  const int lane = tid & 31;
  const int wave = tid >> 5;
  for (int i = tid; i < B_ * HP_; i += 512) { hbuf[i] = (f16_t)0.f; cst[i] = 0.f; }
  __syncthreads();
  const int n_l = lane & 15;
  const int hi8 = (lane >> 4) << 3;
  for (int t = 0; t < T_; ++t) {
    f32x8 harr[5];
    int np = 0;
    for (int p = wave; p < 80; p += 16, ++np) {
      const int mt = p / 20, jt = p % 20;
      const int row0 = mt * 16;
      f32x8 gi = {}, gf = {}, gg = {}, go = {};
      for (int k0 = 0; k0 < HP_; k0 += 32) {
        // group all 5 operand loads so they issue as one clause
        f16x16 a  = load_tile16(hbuf, HP_, row0, k0, lane);
        f16x16 b0 = load_tile16(Whh, HP_, 0 * HP_ + jt * 16, k0, lane);
        f16x16 b1 = load_tile16(Whh, HP_, 1 * HP_ + jt * 16, k0, lane);
        f16x16 b2 = load_tile16(Whh, HP_, 2 * HP_ + jt * 16, k0, lane);
        f16x16 b3 = load_tile16(Whh, HP_, 3 * HP_ + jt * 16, k0, lane);
        gi = __builtin_amdgcn_wmma_f32_16x16x32_f16(false, a, false, b0, (short)0, gi, false, false);
        gf = __builtin_amdgcn_wmma_f32_16x16x32_f16(false, a, false, b1, (short)0, gf, false, false);
        gg = __builtin_amdgcn_wmma_f32_16x16x32_f16(false, a, false, b2, (short)0, gg, false, false);
        go = __builtin_amdgcn_wmma_f32_16x16x32_f16(false, a, false, b3, (short)0, go, false, false);
      }
      const int col = jt * 16 + n_l;
      const float* xwp  = XW + (size_t)t * B_ * GP_;
      const float* xwp1 = xwp + (size_t)B_ * GP_;        // next timestep
      f32x8 hv = {};
#pragma unroll
      for (int r = 0; r < 8; ++r) {
        const int    bI = row0 + hi8 + r;
        const size_t xb = (size_t)bI * GP_ + col;
        if (t + 1 < T_) {   // hide next step's XW latency behind this step's WMMA work
          __builtin_prefetch(xwp1 + xb, 0, 3);
          __builtin_prefetch(xwp1 + xb + HP_, 0, 3);
          __builtin_prefetch(xwp1 + xb + 2 * HP_, 0, 3);
          __builtin_prefetch(xwp1 + xb + 3 * HP_, 0, 3);
        }
        float vi = gi[r] + xwp[xb];
        float vf = gf[r] + xwp[xb + HP_];
        float vg = gg[r] + xwp[xb + 2 * HP_];
        float vo = go[r] + xwp[xb + 3 * HP_];
        float si = 1.f / (1.f + expf(-vi));
        float sf = 1.f / (1.f + expf(-vf));
        float so = 1.f / (1.f + expf(-vo));
        float c  = sf * cst[bI * HP_ + col] + si * tanhf(vg);
        cst[bI * HP_ + col] = c;
        float h = so * tanhf(c);
        hv[r] = h;
        h_all[((size_t)bI * T_ + t) * HP_ + col] = (f16_t)h;
      }
      harr[np] = hv;
    }
    __syncthreads();     // all reads of hbuf (h_prev) complete
    np = 0;
    for (int p = wave; p < 80; p += 16, ++np) {
      const int mt = p / 20, jt = p % 20;
      const int col  = jt * 16 + n_l;
      const int rowb = mt * 16 + hi8;
#pragma unroll
      for (int r = 0; r < 8; ++r)
        hbuf[(rowb + r) * HP_ + col] = (f16_t)harr[np][r];
    }
    __syncthreads();     // h_t visible before next step's GEMM
  }
}

// ---------------- output projection ----------------
__global__ void k_proj(const f16_t* __restrict__ H, const float* __restrict__ Wout,
                       const float* __restrict__ bout, float* __restrict__ feats) {
  int idx = blockIdx.x * blockDim.x + threadIdx.x;
  int k  = idx & 15;
  int bt = idx >> 4;
  if (bt >= B_ * T_ || k >= K_) return;
  const f16_t* h = H + (size_t)bt * LDA_;
  const float* w = Wout + (size_t)k * (2 * H_);
  float s = bout[k];
  for (int c = 0; c < 2 * H_; ++c) s += (float)h[c] * w[c];
  feats[(size_t)bt * K_ + k] = s;
}

// ---------------- Viterbi (one lane per batch element) ----------------
__global__ void k_viterbi(const float* __restrict__ feats, const int* __restrict__ lens,
                          const float* __restrict__ trans, int* __restrict__ bp,
                          float* __restrict__ out) {
  int b = threadIdx.x;
  if (b >= B_) return;
  int len = lens[b];
  float tr[K_][K_];
  for (int i = 0; i < K_; ++i)
    for (int j = 0; j < K_; ++j) tr[i][j] = trans[i * K_ + j];
  float fv[K_];
  for (int k = 0; k < K_; ++k) fv[k] = -10.f;
  fv[VSTART_] = 0.f;
  for (int t = 0; t < T_; ++t) {
    float vit[K_];
    for (int kn = 0; kn < K_; ++kn) {
      float best = fv[0] + tr[kn][0]; int bi = 0;
      for (int kp = 1; kp < K_; ++kp) {
        float v = fv[kp] + tr[kn][kp];
        if (v > best) { best = v; bi = kp; }    // first-max like jnp.argmax
      }
      vit[kn] = best;
      bp[((size_t)t * B_ + b) * K_ + kn] = bi;
    }
    if (t < len)
      for (int kn = 0; kn < K_; ++kn)
        fv[kn] = vit[kn] + feats[((size_t)b * T_ + t) * K_ + kn];
  }
  float best = fv[0] + tr[VSTOP_][0]; int btag = 0;
  for (int k = 1; k < K_; ++k) {
    float v = fv[k] + tr[VSTOP_][k];
    if (v > best) { best = v; btag = k; }
  }
  out[b] = best;
  float* tags = out + B_;
  int tag = 0;
  for (int t = T_ - 1; t >= 0; --t) {
    int tnext = (t + 1 < T_) ? (t + 1) : (T_ - 1);
    int step_tag = bp[((size_t)tnext * B_ + b) * K_ + tag];
    int nt = (t == len - 1) ? btag : step_tag;
    if (t >= len) nt = 0;
    tags[(size_t)b * T_ + t] = (t < len) ? (float)nt : -1.f;
    tag = nt;
  }
}

// ---------------- host orchestration ----------------
extern "C" void kernel_launch(void* const* d_in, const int* in_sizes, int n_in,
                              void* d_out, int out_size, void* d_ws, size_t ws_size,
                              hipStream_t stream) {
  (void)in_sizes; (void)n_in; (void)out_size; (void)ws_size;
  const int*   x    = (const int*)d_in[0];
  const int*   lens = (const int*)d_in[1];
  const float* emb  = (const float*)d_in[2];
  const float* Wih[2][2] = {{(const float*)d_in[3],  (const float*)d_in[6]},
                            {(const float*)d_in[9],  (const float*)d_in[12]}};
  const float* Whh[2][2] = {{(const float*)d_in[4],  (const float*)d_in[7]},
                            {(const float*)d_in[10], (const float*)d_in[13]}};
  const float* bia[2][2] = {{(const float*)d_in[5],  (const float*)d_in[8]},
                            {(const float*)d_in[11], (const float*)d_in[14]}};
  const float* Wout  = (const float*)d_in[15];
  const float* bout  = (const float*)d_in[16];
  const float* trans = (const float*)d_in[17];

  char* w = (char*)d_ws;
  f16_t* bufA   = (f16_t*)(w + 0);            // [B,T,640]
  f16_t* bufRev = (f16_t*)(w + 20971520);     // [B,T,640]
  f16_t* bufCat = (f16_t*)(w + 41943040);     // [B,T,640]
  f16_t* hF     = (f16_t*)(w + 62914560);     // [B,T,320]
  f16_t* hB     = (f16_t*)(w + 73400320);     // [B,T,320]
  float* XWb    = (float*)(w + 83886080);     // [T,B,1280] f32
  f16_t* WIH    = (f16_t*)(w + 167772160);    // [1280,640]
  f16_t* WHHp   = (f16_t*)(w + 169410560);    // [1280,320]
  float* BIAS   = (float*)(w + 170229760);    // [1280]
  float* FEAT   = (float*)(w + 170234880);    // [B,T,11]
  int*   BP     = (int*)  (w + 170955776);    // [T,B,11]
  float* CST    = (float*)(w + 171676672);    // [B,320] cell-state scratch

  dim3 blk(256);
  const int nElem = B_ * T_ * LDA_;
  k_gather_embed<<<(nElem + 255) / 256, blk, 0, stream>>>(x, emb, bufA);

  const f16_t* cur = bufA;
  for (int l = 0; l < 2; ++l) {
    const int indim = l ? 2 * H_ : H_;
    const int kdim  = l ? LDA_ : HP_;
    f16_t* dst = l ? bufA : bufCat;
    for (int d = 0; d < 2; ++d) {
      const f16_t* src = cur;
      if (d == 1) {
        k_rev<<<(nElem + 255) / 256, blk, 0, stream>>>(cur, bufRev, lens);
        src = bufRev;
      }
      k_prep_wih<<<(GP_ * LDA_ + 255) / 256, blk, 0, stream>>>(Wih[l][d], WIH, indim);
      k_prep_whh<<<(GP_ * HP_ + 255) / 256, blk, 0, stream>>>(Whh[l][d], WHHp);
      k_prep_bias<<<(GP_ + 255) / 256, blk, 0, stream>>>(bia[l][d], BIAS);
      k_gemm_xw<<<2560, blk, 0, stream>>>(src, kdim, WIH, BIAS, XWb);
      k_lstm_scan<<<1, 512, 0, stream>>>(XWb, WHHp, CST, d ? hB : hF);
    }
    k_concat<<<(nElem + 255) / 256, blk, 0, stream>>>(hF, hB, lens, dst);
    cur = dst;
  }
  k_proj<<<(B_ * T_ * 16 + 255) / 256, blk, 0, stream>>>(cur, Wout, bout, FEAT);
  k_viterbi<<<1, 64, 0, stream>>>(FEAT, lens, trans, BP, (float*)d_out);
}
